// CAE_41051297415282
// MI455X (gfx1250) — matrix operand order repaired and checked by reading, not compiled, not run on hardware
//
#include <hip/hip_runtime.h>
#include <hip/hip_bf16.h>

typedef __attribute__((ext_vector_type(16))) _Float16 v16h;
typedef __attribute__((ext_vector_type(8)))  _Float16 v8h;
typedef __attribute__((ext_vector_type(8)))  float    v8f;

#define NPIX   262144      // 512*512
#define NQUAD  65536       // NPIX/4
#define CUBE_N 31457280    // 120*512*512
#define VLIMF  600.0f
#define PI_F   3.14159265358979323846f

// ---------------------------------------------------------------------------
// Stage 1: pack x [120][64][64] f32 -> xt [66][66][128] f16 channel-last,
// zero spatial border (pad=1) and zero channels 120..127. After this, every
// A-fragment K-run of the implicit GEMM is contiguous & 16B aligned.
// ---------------------------------------------------------------------------
__global__ void pack_x(const float* __restrict__ x, _Float16* __restrict__ xt) {
  int idx = blockIdx.x * 256 + threadIdx.x;     // one uint (2 halfs) per thread
  if (idx >= 66 * 66 * 64) return;
  int cpair = idx & 63;                          // c = cpair*2
  int sp = idx >> 6;                             // 0..4355
  int gy = sp / 66, gx = sp - gy * 66;
  int c = cpair * 2;
  float f0 = 0.0f, f1 = 0.0f;
  int iy = gy - 1, ix = gx - 1;
  if ((unsigned)iy < 64u && (unsigned)ix < 64u) {
    int pos = iy * 64 + ix;
    if (c < 120)     f0 = x[c * 4096 + pos];
    if (c + 1 < 120) f1 = x[(c + 1) * 4096 + pos];
  }
  unsigned u = (unsigned)__builtin_bit_cast(unsigned short, (_Float16)f0) |
               ((unsigned)__builtin_bit_cast(unsigned short, (_Float16)f1) << 16);
  ((unsigned*)xt)[idx] = u;
}

// ---------------------------------------------------------------------------
// Stage 2: pack w0 [16][120][3][3] f32 -> pb: exact B-fragment records, f16.
// Record (chunk, lane) = 16 halfs (32B). chunk = tap*4 + cq (K = tap*128 + c).
// B 32x16 layout: lane n%16 = column; lanes<16 hold K cc+{0..7,16..23}, lanes>=16
// hold K cc+{8..15,24..31} (2 halfs per VGPR).
// ---------------------------------------------------------------------------
__global__ void pack_b(const float* __restrict__ w0, _Float16* __restrict__ pb) {
  int rec = blockIdx.x * 256 + threadIdx.x;     // 0..1151
  if (rec >= 1152) return;
  int ch = rec >> 5, lane = rec & 31;
  int half = lane >> 4, ncol = lane & 15;
  int tap = ch >> 2, cc = (ch & 3) << 5;
  int kb = cc + half * 8;
  v16h vv;
#pragma unroll
  for (int i = 0; i < 16; ++i) {
    int c = kb + i + (i >= 8 ? 8 : 0);
    float v = (c < 120) ? w0[ncol * 1080 + c * 9 + tap] : 0.0f;
    vv[i] = (_Float16)v;
  }
  *(v16h*)(pb + ((size_t)rec << 4)) = vv;
}

// ---------------------------------------------------------------------------
// conv0 implicit GEMM with WMMA: M=4096 (64x64 spatial), N=16, K=9*128.
// One wave computes TWO 16x16 D tiles (shared B). Inner loop is pure
// b128 loads + 2x v_wmma_f32_16x16x32_f16 per 32-K chunk.
// A 16x32 f16 layout: lanes 0-15: M=lane,  K {0..7,16..23};
//                     lanes16-31: M=l-16,  K {8..15,24..31}.
// With xt channel-last these runs are p[cq*4+half] and p[cq*4+half+2].
// D: vgpr r -> M = r + 8*(lane>=16); N = lane%16.
// ---------------------------------------------------------------------------
__global__ void conv0_wmma(const _Float16* __restrict__ xt, const _Float16* __restrict__ pb,
                           const float* __restrict__ b0, float* __restrict__ out) {
  const int lane = threadIdx.x & 31;
  const int wave = threadIdx.x >> 5;
  const int pair = blockIdx.x * 4 + wave;       // 0..127, two 16-row tiles each
  const int mbase0 = pair * 32, mbase1 = mbase0 + 16;
  const int half = lane >> 4;
  const int mrow = lane & 15;
  const int ncol = lane & 15;
  const int pos0 = mbase0 + mrow, pos1 = mbase1 + mrow;
  const int py0 = pos0 >> 6, px0 = pos0 & 63;
  const int py1 = pos1 >> 6, px1 = pos1 & 63;

  v8f acc0 = {}, acc1 = {};

#pragma unroll
  for (int tap = 0; tap < 9; ++tap) {
    const int ky = tap / 3, kx = tap - (tap / 3) * 3;   // folds at compile time
    const v8h* p0 = (const v8h*)(xt + ((py0 + ky) * 66 + (px0 + kx)) * 128);
    const v8h* p1 = (const v8h*)(xt + ((py1 + ky) * 66 + (px1 + kx)) * 128);
    const v16h* pbt = (const v16h*)(pb + (((size_t)tap * 4 * 32 + lane) << 4));
#pragma unroll
    for (int cq = 0; cq < 4; ++cq) {
      const int i0 = cq * 4 + half;
      v8h a0lo = p0[i0], a0hi = p0[i0 + 2];
      v8h a1lo = p1[i0], a1hi = p1[i0 + 2];
      v16h bfrag = pbt[cq * 32];
      v16h a0 = __builtin_shufflevector(a0lo, a0hi, 0,1,2,3,4,5,6,7,8,9,10,11,12,13,14,15);
      v16h a1 = __builtin_shufflevector(a1lo, a1hi, 0,1,2,3,4,5,6,7,8,9,10,11,12,13,14,15);
      acc0 = __builtin_amdgcn_wmma_f32_16x16x32_f16(false, a0, false, bfrag,
                                                    (short)0, acc0, false, false);
      acc1 = __builtin_amdgcn_wmma_f32_16x16x32_f16(false, a1, false, bfrag,
                                                    (short)0, acc1, false, false);
    }
  }
  float bias = b0[ncol];
#pragma unroll
  for (int r = 0; r < 8; ++r) {
    int M = r + 8 * half;
    out[ncol * 4096 + mbase0 + M] = acc0[r] + bias;   // no relu on conv0
    out[ncol * 4096 + mbase1 + M] = acc1[r] + bias;
  }
}

// 2x2 max pool of conv0 output: [16][64][64] -> [16][32][32]
__global__ void pool0(const float* __restrict__ in, float* __restrict__ out) {
  int idx = blockIdx.x * blockDim.x + threadIdx.x;   // 16384 total
  if (idx >= 16384) return;
  int ox = idx & 31, oy = (idx >> 5) & 31, ch = idx >> 10;
  const float* p = in + ch * 4096;
  int y = oy * 2, x = ox * 2;
  float m = fmaxf(fmaxf(p[y * 64 + x], p[y * 64 + x + 1]),
                  fmaxf(p[(y + 1) * 64 + x], p[(y + 1) * 64 + x + 1]));
  out[idx] = m;
}

// Generic fused 3x3 conv (pad 1) + relu + 2x2 max pool. One thread per pooled out.
__global__ void conv_relu_pool(const float* __restrict__ in, const float* __restrict__ w,
                               const float* __restrict__ b, float* __restrict__ out,
                               int Cin, int Hin, int Cout) {
  int Hout = Hin >> 1;
  int idx = blockIdx.x * blockDim.x + threadIdx.x;
  int total = Cout * Hout * Hout;
  if (idx >= total) return;
  int ox = idx % Hout, t = idx / Hout;
  int oy = t % Hout, co = t / Hout;
  float m = -3.0e38f;
#pragma unroll
  for (int dy = 0; dy < 2; ++dy)
#pragma unroll
    for (int dx = 0; dx < 2; ++dx) {
      int y = oy * 2 + dy, x = ox * 2 + dx;
      float accv = b[co];
      for (int ci = 0; ci < Cin; ++ci) {
        const float* ip = in + ci * Hin * Hin;
        const float* wp = w + (co * Cin + ci) * 9;
#pragma unroll
        for (int ky = 0; ky < 3; ++ky) {
          int iy = y + ky - 1;
          if ((unsigned)iy >= (unsigned)Hin) continue;
#pragma unroll
          for (int kx = 0; kx < 3; ++kx) {
            int ix = x + kx - 1;
            if ((unsigned)ix >= (unsigned)Hin) continue;
            accv += ip[iy * Hin + ix] * wp[ky * 3 + kx];
          }
        }
      }
      m = fmaxf(m, accv);
    }
  out[idx] = fmaxf(m, 0.0f);   // relu(pool(x)) == pool(relu(x))
}

// Fully connected: out[i] = act(w[i,:] . in + b[i]); act 0=none 1=relu 2=hardtanh
__global__ void fc(const float* __restrict__ in, const float* __restrict__ w,
                   const float* __restrict__ b, float* __restrict__ out,
                   int In, int Out, int act) {
  int i = blockIdx.x * blockDim.x + threadIdx.x;
  if (i >= Out) return;
  float accv = b[i];
  const float* wp = w + (size_t)i * In;
  for (int k = 0; k < In; ++k) accv += wp[k] * in[k];
  if (act == 1) accv = fmaxf(accv, 0.0f);
  else if (act == 2) accv = fminf(fmaxf(accv, -1.0f), 1.0f);
  out[i] = accv;
}

// Single-thread: latents -> derived params; init reduction + present state.
__global__ void make_params(const float* __restrict__ lat, float* __restrict__ params,
                            unsigned int* __restrict__ rrmm, unsigned int* __restrict__ pm) {
  if (threadIdx.x == 0 && blockIdx.x == 0) {
    float pos = atan2f(lat[0], lat[1]) + PI_F;
    float inc = (lat[2] + 1.0f) * (PI_F * 0.5f - 5.0f) * 0.5f + 5.0f;
    float a   = ((lat[3] + 1.0f) * (0.5f - 0.1f) * 0.5f + 0.1f) * 256.0f;
    float ah  = ((lat[4] + 1.0f) * (0.1f - 0.01f) * 0.5f + 0.01f) * 256.0f;
    float Vh  = (lat[5] + 1.0f) * (500.0f - 50.0f) * 0.5f + 50.0f;
    params[0] = sinf(pos);
    params[1] = cosf(pos);
    params[2] = 1.0f / sinf(PI_F * 0.5f - inc);
    params[3] = a;
    params[4] = ah;
    params[5] = Vh;
    params[6] = sinf(inc);
    rrmm[0] = 0x7F800000u;   // +inf bits (min slot); rr >= 0 so uint order == float order
    rrmm[1] = 0u;            // max slot
    pm[0] = pm[1] = pm[2] = pm[3] = 0u;
  }
}

// Global min/max of rr (as uint bits; rr non-negative -> uint order == float order).
__global__ void rr_minmax(const float* __restrict__ xx, const float* __restrict__ yy,
                          const float* __restrict__ params, unsigned int* __restrict__ rrmm) {
  __shared__ unsigned int smn[256], smx[256];
  int p = blockIdx.x * 256 + threadIdx.x;
  float sp = params[0], cp = params[1], inv = params[2];
  float X = xx[p], Y = yy[p];
  float xt = -X * sp + Y * cp;
  float yt = (X * cp + Y * sp) * inv;
  float rr = sqrtf(xt * xt + yt * yt);
  unsigned int u = __float_as_uint(rr);
  smn[threadIdx.x] = u;
  smx[threadIdx.x] = u;
  for (int s = 128; s > 0; s >>= 1) {
    __syncthreads();
    if ((int)threadIdx.x < s) {
      unsigned a0 = smn[threadIdx.x], a1 = smn[threadIdx.x + s];
      smn[threadIdx.x] = a0 < a1 ? a0 : a1;
      unsigned c0 = smx[threadIdx.x], c1 = smx[threadIdx.x + s];
      smx[threadIdx.x] = c0 > c1 ? c0 : c1;
    }
  }
  __syncthreads();
  if (threadIdx.x == 0) {
    atomicMin(&rrmm[0], smn[0]);
    atomicMax(&rrmm[1], smx[0]);
  }
}

// Per-pixel sb, los velocity, present-bin mask.
__global__ void sbvel(const float* __restrict__ xx, const float* __restrict__ yy,
                      const float* __restrict__ params, const unsigned int* __restrict__ rrmm,
                      unsigned int* __restrict__ pm,
                      float* __restrict__ vout, float* __restrict__ sbout) {
  __shared__ unsigned int lpm[4];
  if (threadIdx.x < 4) lpm[threadIdx.x] = 0u;
  __syncthreads();
  int p = blockIdx.x * 256 + threadIdx.x;
  float sp = params[0], cp = params[1], inv = params[2];
  float a = params[3], ah = params[4], Vh = params[5], si = params[6];
  float X = xx[p], Y = yy[p];
  float xt = -X * sp + Y * cp;
  float yt = (X * cp + Y * sp) * inv;
  float rr = sqrtf(xt * xt + yt * yt);
  float sbmin = expf(-__uint_as_float(rrmm[1]) / a);   // exp monotone: min at rr_max
  float sbmax = expf(-__uint_as_float(rrmm[0]) / a);
  float sb = (expf(-rr / a) - sbmin) / (sbmax - sbmin);
  float vel = sqrtf(Vh * Vh * (1.0f - ah / rr * atanf(rr / ah)));
  vel = vel * cosf(atan2f(yt, xt)) * si;
  if (vel < -VLIMF || vel > VLIMF) vel = 0.0f;
  vout[p] = vel;
  sbout[p] = sb;
  int ibin = (int)floorf(vel / 10.0f) + 60;
  if (ibin >= 0 && ibin < 120)
    atomicOr(&lpm[ibin >> 5], 1u << (ibin & 31));
  __syncthreads();
  if (threadIdx.x < 4 && lpm[threadIdx.x]) atomicOr(&pm[threadIdx.x], lpm[threadIdx.x]);
}

// Cube fill, vectorized: one thread per 4 consecutive pixels, b128 loads/stores.
// 126 MB of stores -> this kernel IS the roofline cost (~5.4us @ 23.3 TB/s).
__global__ void fill4(const float4* __restrict__ v4, const float4* __restrict__ sb4,
                      const float4* __restrict__ cube4, const unsigned int* __restrict__ pm,
                      float4* __restrict__ out4) {
  int q = blockIdx.x * 256 + threadIdx.x;    // 0..65535
  float4 vv = v4[q], ss = sb4[q];
  int b0 = (int)floorf(vv.x / 10.0f) + 60;
  int b1 = (int)floorf(vv.y / 10.0f) + 60;
  int b2 = (int)floorf(vv.z / 10.0f) + 60;
  int b3 = (int)floorf(vv.w / 10.0f) + 60;
  unsigned m0 = pm[0], m1 = pm[1], m2 = pm[2], m3 = pm[3];
  for (int w = 0; w < 120; ++w) {
    unsigned mw = (w < 32) ? m0 : (w < 64) ? m1 : (w < 96) ? m2 : m3;
    size_t o = (size_t)w * NQUAD + q;
    float4 val;
    if ((mw >> (w & 31)) & 1u) {             // uniform branch per plane
      val.x = (b0 == w) ? ss.x : 0.0f;
      val.y = (b1 == w) ? ss.y : 0.0f;
      val.z = (b2 == w) ? ss.z : 0.0f;
      val.w = (b3 == w) ? ss.w : 0.0f;
    } else {
      val = cube4[o];                        // keep input cube where bin absent
    }
    out4[o] = val;
  }
}

extern "C" void kernel_launch(void* const* d_in, const int* in_sizes, int n_in,
                              void* d_out, int out_size, void* d_ws, size_t ws_size,
                              hipStream_t stream) {
  (void)in_sizes; (void)n_in; (void)out_size; (void)ws_size;
  const float* x    = (const float*)d_in[0];
  const float* xx   = (const float*)d_in[1];
  const float* yy   = (const float*)d_in[2];
  const float* cube = (const float*)d_in[3];
  const float* w0 = (const float*)d_in[4];   const float* b0 = (const float*)d_in[5];
  const float* w1 = (const float*)d_in[6];   const float* b1 = (const float*)d_in[7];
  const float* w2 = (const float*)d_in[8];   const float* b2 = (const float*)d_in[9];
  const float* w3 = (const float*)d_in[10];  const float* b3 = (const float*)d_in[11];
  const float* wl1 = (const float*)d_in[12]; const float* bl1 = (const float*)d_in[13];
  const float* wl2 = (const float*)d_in[14]; const float* bl2 = (const float*)d_in[15];
  const float* wl3 = (const float*)d_in[16]; const float* bl3 = (const float*)d_in[17];
  float* out = (float*)d_out;

  float* wsf      = (float*)d_ws;
  float* conv0buf = wsf;             // 65536
  float* act0     = wsf + 65536;     // 16384
  float* act1     = wsf + 81920;     // 8192
  float* act2     = wsf + 90112;     // 4096
  float* act3     = wsf + 94208;     // 2048
  float* fc1o     = wsf + 96256;     // 1024
  float* fc2o     = wsf + 97280;     // 256
  float* lat      = wsf + 97536;     // 6 (+pad)
  float* params   = wsf + 97544;     // 8
  unsigned int* rrmm = (unsigned int*)(wsf + 97552);  // [2]
  unsigned int* pm   = rrmm + 2;                      // [4]
  _Float16* xtb = (_Float16*)(wsf + 98304);   // 66*66*128 = 557568 halfs (16B aligned)
  _Float16* pbb = (_Float16*)(wsf + 377088);  // 1152*16   =  18432 halfs

  float* vout  = out + CUBE_N;
  float* sbout = out + CUBE_N + NPIX;

  // ---- encoder ----
  pack_x<<<1089, 256, 0, stream>>>(x, xtb);
  pack_b<<<5, 256, 0, stream>>>(w0, pbb);
  conv0_wmma<<<32, 128, 0, stream>>>(xtb, pbb, b0, conv0buf);
  pool0<<<64, 256, 0, stream>>>(conv0buf, act0);
  conv_relu_pool<<<32, 256, 0, stream>>>(act0, w1, b1, act1, 16, 32, 32);
  conv_relu_pool<<<16, 256, 0, stream>>>(act1, w2, b2, act2, 32, 16, 64);
  conv_relu_pool<<<8, 256, 0, stream>>>(act2, w3, b3, act3, 64, 8, 128);
  fc<<<4, 256, 0, stream>>>(act3, wl1, bl1, fc1o, 2048, 1024, 1);
  fc<<<1, 256, 0, stream>>>(fc1o, wl2, bl2, fc2o, 1024, 256, 1);
  fc<<<1, 32, 0, stream>>>(fc2o, wl3, bl3, lat, 256, 6, 2);

  // ---- cube maker ----
  make_params<<<1, 1, 0, stream>>>(lat, params, rrmm, pm);
  rr_minmax<<<1024, 256, 0, stream>>>(xx, yy, params, rrmm);
  sbvel<<<1024, 256, 0, stream>>>(xx, yy, params, rrmm, pm, vout, sbout);
  fill4<<<256, 256, 0, stream>>>((const float4*)vout, (const float4*)sbout,
                                 (const float4*)cube, pm, (float4*)out);
}
